// AttGCN_47064251630163
// MI455X (gfx1250) — compile-verified, matrix operand order checked by reading, not run on hardware
//
#include <hip/hip_runtime.h>
#include <math.h>

// ---------------------------------------------------------------------------
// FeaStConv U-Net for MI455X (gfx1250, wave32, WMMA).
// Node-hoisted formulation: per block
//   M = bf16(X) @ bf16([W | u])   (WMMA 16x16x32 bf16 -> f32)
//   per-edge: q = softmax(Mu[src]-Mu[dst]+c); scatter sum_h q_h * MW[src,h,:]
//   finalize: mean + bias (+ LayerNorm) + skip (+ ReLU)
// ---------------------------------------------------------------------------

typedef __attribute__((ext_vector_type(16))) __bf16 v16bf;
typedef __attribute__((ext_vector_type(8)))  float  v8f;

union V16U { uint4 q[2]; v16bf v; };

static __device__ __forceinline__ unsigned short f2bf(float f) {
    unsigned u = __float_as_uint(f);
    u += 0x7FFFu + ((u >> 16) & 1u);
    return (unsigned short)(u >> 16);
}

// ----- fp32 [rows x cols] -> bf16 [rowsPad x kpad], zero padded -------------
__global__ void k_f32_to_bf16_pad(const float* __restrict__ src,
                                  unsigned short* __restrict__ dst,
                                  int rows, int cols, int rowsPad, int kpad) {
    int i = blockIdx.x * blockDim.x + threadIdx.x;
    int total = rowsPad * kpad;
    if (i >= total) return;
    int r = i / kpad, k = i - r * kpad;
    float v = (r < rows && k < cols) ? src[(size_t)r * cols + k] : 0.0f;
    dst[i] = f2bf(v);
}

// ----- pack W (KxcoutW) [+ u (KxH)] into WMMA B-fragment panels -------------
// layout: Bp[(((ks*ctiles)+ct)*32 + lane)*16 + e], lane holds col n = ct*16+(lane&15),
// element e -> k = ks*32 + ((lane&16)?8:0) + ((e&8)?16:0) + (e&7)
__global__ void k_pack_b(const float* __restrict__ W, const float* __restrict__ u,
                         unsigned short* __restrict__ Bp,
                         int K, int kpad, int coutW, int H, int ctiles) {
    int i = blockIdx.x * blockDim.x + threadIdx.x;
    int total = kpad * ctiles * 16;
    if (i >= total) return;
    int e = i & 15; int t = i >> 4;
    int lane = t & 31; t >>= 5;
    int ct = t % ctiles; int ks = t / ctiles;
    int k = ks * 32 + ((lane & 16) ? 8 : 0) + ((e & 8) ? 16 : 0) + (e & 7);
    int n = ct * 16 + (lane & 15);
    float v = 0.0f;
    if (k < K) {
        if (n < coutW)                 v = W[(size_t)k * coutW + n];
        else if (u && n < coutW + H)   v = u[(size_t)k * H + (n - coutW)];
    }
    Bp[i] = f2bf(v);
}

// ----- WMMA GEMM: Out[n x (ctiles*16)] = A[nPad x kpad] * Bp ----------------
__global__ void k_wmma_gemm(const unsigned short* __restrict__ A,
                            const unsigned short* __restrict__ Bp,
                            float* __restrict__ Out,
                            int nrows, int kpad, int ctiles) {
    int wave = blockIdx.x * (blockDim.x >> 5) + (threadIdx.x >> 5);
    int lane = threadIdx.x & 31;
    int tiles = ((nrows + 15) >> 4) * ctiles;
    if (wave >= tiles) return;
    int rt = wave / ctiles, ct = wave - rt * ctiles;

    const unsigned short* arow =
        A + (size_t)(rt * 16 + (lane & 15)) * kpad + ((lane >> 4) & 1) * 8;
    const unsigned short* bptr = Bp + ((size_t)ct * 32 + lane) * 16;
    size_t bstep = (size_t)ctiles * 512;

    v8f acc = {0.f, 0.f, 0.f, 0.f, 0.f, 0.f, 0.f, 0.f};
    int ksteps = kpad >> 5;
    for (int ks = 0; ks < ksteps; ++ks) {
        V16U a, b;
        a.q[0] = *reinterpret_cast<const uint4*>(arow);
        a.q[1] = *reinterpret_cast<const uint4*>(arow + 16);
        b.q[0] = *reinterpret_cast<const uint4*>(bptr);
        b.q[1] = *reinterpret_cast<const uint4*>(bptr + 8);
        acc = __builtin_amdgcn_wmma_f32_16x16x32_bf16(
            false, a.v, false, b.v, (short)0, acc, false, false);
        arow += 32;
        bptr += bstep;
    }
    int ldo = ctiles * 16;
    int mbase = rt * 16 + ((lane & 16) ? 8 : 0);
    int n = ct * 16 + (lane & 15);
#pragma unroll
    for (int r = 0; r < 8; ++r) {
        int m = mbase + r;
        if (m < nrows) Out[(size_t)m * ldo + n] = acc[r];
    }
}

// ----- per-edge softmax-attention scatter (one wave per edge) ---------------
__global__ void k_edge_aggregate(const float* __restrict__ M,
                                 const int* __restrict__ src,
                                 const int* __restrict__ dst,
                                 const float* __restrict__ cv,
                                 float* __restrict__ acc, float* __restrict__ cnt,
                                 int E, int C, int hcoff, int ldm) {
    int e = blockIdx.x * (blockDim.x >> 5) + (threadIdx.x >> 5);
    if (e >= E) return;
    int lane = threadIdx.x & 31;
    int s = src[e], d = dst[e];
    const float* Ms = M + (size_t)s * ldm;
    const float* Md = M + (size_t)d * ldm;
    float a0 = Ms[hcoff + 0] - Md[hcoff + 0] + cv[0];
    float a1 = Ms[hcoff + 1] - Md[hcoff + 1] + cv[1];
    float a2 = Ms[hcoff + 2] - Md[hcoff + 2] + cv[2];
    float a3 = Ms[hcoff + 3] - Md[hcoff + 3] + cv[3];
    float mx = fmaxf(fmaxf(a0, a1), fmaxf(a2, a3));
    float q0 = __expf(a0 - mx), q1 = __expf(a1 - mx);
    float q2 = __expf(a2 - mx), q3 = __expf(a3 - mx);
    float inv = 1.0f / (q0 + q1 + q2 + q3);
    q0 *= inv; q1 *= inv; q2 *= inv; q3 *= inv;
    float* ad = acc + (size_t)d * C;
    for (int c = lane; c < C; c += 32) {
        float v = q0 * Ms[c] + q1 * Ms[C + c] + q2 * Ms[2 * C + c] + q3 * Ms[3 * C + c];
        atomicAdd(ad + c, v);
    }
    if (lane == 0) atomicAdd(cnt + d, 1.0f);
}

// ----- finalize: mean + bias (+LN) + skip (+ReLU); one wave per node --------
__global__ void k_finalize(const float* __restrict__ acc, const float* __restrict__ cnt,
                           const float* __restrict__ bias,
                           const float* __restrict__ g, const float* __restrict__ be,
                           const float* __restrict__ skip, int skipStride,
                           const float* __restrict__ sb,
                           float* __restrict__ out, int N, int C, int relu) {
    int n = blockIdx.x * (blockDim.x >> 5) + (threadIdx.x >> 5);
    if (n >= N) return;
    int lane = threadIdx.x & 31;
    float rinv = 1.0f / fmaxf(cnt[n], 1.0f);
    float vals[8];
    int nk = (C + 31) >> 5;
    for (int j = 0; j < nk; ++j) {
        int c = lane + (j << 5);
        vals[j] = (c < C) ? (acc[(size_t)n * C + c] * rinv + bias[c]) : 0.0f;
    }
    if (g) {
        float s = 0.0f;
        for (int j = 0; j < nk; ++j) s += vals[j];
        for (int o = 16; o; o >>= 1) s += __shfl_xor(s, o, 32);
        float mu = s / (float)C;
        float v = 0.0f;
        for (int j = 0; j < nk; ++j) {
            int c = lane + (j << 5);
            if (c < C) { float dd = vals[j] - mu; v += dd * dd; }
        }
        for (int o = 16; o; o >>= 1) v += __shfl_xor(v, o, 32);
        float rs = rsqrtf(v / (float)C + 1e-5f);
        for (int j = 0; j < nk; ++j) {
            int c = lane + (j << 5);
            if (c < C) vals[j] = (vals[j] - mu) * rs * g[c] + be[c];
        }
    }
    for (int j = 0; j < nk; ++j) {
        int c = lane + (j << 5);
        if (c < C) {
            float sk = skip[(size_t)n * skipStride + c] + (sb ? sb[c] : 0.0f);
            float h = vals[j] + sk;
            if (relu) h = fmaxf(h, 0.0f);
            out[(size_t)n * C + c] = h;
        }
    }
}

// ----- pooling / unpooling ---------------------------------------------------
__global__ void k_pool_acc(const float* __restrict__ x, const int* __restrict__ cl,
                           float* __restrict__ acc, float* __restrict__ cnt,
                           int N, int C) {
    int i = blockIdx.x * blockDim.x + threadIdx.x;
    if (i >= N * C) return;
    int n = i / C, c = i - n * C;
    int t = cl[n];
    atomicAdd(&acc[(size_t)t * C + c], x[i]);
    if (c == 0) atomicAdd(&cnt[t], 1.0f);
}

__global__ void k_pool_div(const float* __restrict__ acc, const float* __restrict__ cnt,
                           float* __restrict__ out, int N, int C) {
    int i = blockIdx.x * blockDim.x + threadIdx.x;
    if (i >= N * C) return;
    int n = i / C;
    out[i] = acc[i] / fmaxf(cnt[n], 1.0f);
}

__global__ void k_unpool_concat(const float* __restrict__ coarse, const int* __restrict__ cl,
                                const float* __restrict__ fine, float* __restrict__ out,
                                int N, int C) {
    int i = blockIdx.x * blockDim.x + threadIdx.x;
    int C2 = 2 * C;
    if (i >= N * C2) return;
    int n = i / C2, c = i - n * C2;
    out[i] = (c < C) ? coarse[(size_t)cl[n] * C + c] : fine[(size_t)n * C + (c - C)];
}

// ---------------------------------------------------------------------------
// Host-side orchestration
// ---------------------------------------------------------------------------
namespace {

enum { Cc = 140, Hh = 4, N0 = 20000, N1 = 5000, N2 = 1250,
       E0 = 320000, E1 = 80000, E2 = 20000 };

struct Ctx {
    hipStream_t stream;
    unsigned short *Abf, *Bp, *Bp2;
    float *M, *Sk, *acc, *cnt;
};

static inline int cdiv(int a, int b) { return (a + b - 1) / b; }

static void run_conv(const Ctx& c,
                     const float* X, int N, int Cin, int kpad,
                     const float* W, const float* u, const float* cv,
                     const float* bias, const float* g, const float* be,
                     int Cout,
                     const float* sW, const float* sb,
                     const int* esrc, const int* edst, int E,
                     float* out, int relu) {
    hipStream_t s = c.stream;
    int Npad = cdiv(N, 16) * 16;
    // 1. X -> bf16 padded
    k_f32_to_bf16_pad<<<cdiv(Npad * kpad, 256), 256, 0, s>>>(X, c.Abf, N, Cin, Npad, kpad);
    // 2. pack [W | u]
    int coutW = Hh * Cout;
    int ctiles = cdiv(coutW + Hh, 16);
    int ldm = ctiles * 16;
    k_pack_b<<<cdiv(kpad * ctiles * 16, 256), 256, 0, s>>>(W, u, c.Bp, Cin, kpad, coutW, Hh, ctiles);
    // 3. M = X @ [W|u]
    int tiles = (Npad / 16) * ctiles;
    k_wmma_gemm<<<cdiv(tiles, 4), 128, 0, s>>>(c.Abf, c.Bp, c.M, N, kpad, ctiles);
    // 4. zero accumulators
    hipMemsetAsync(c.acc, 0, (size_t)N * Cout * sizeof(float), s);
    hipMemsetAsync(c.cnt, 0, (size_t)N * sizeof(float), s);
    // 5. edge softmax scatter
    k_edge_aggregate<<<cdiv(E, 8), 256, 0, s>>>(c.M, esrc, edst, cv, c.acc, c.cnt, E, Cout, coutW, ldm);
    // 6. skip
    const float* skipSrc; int skipStride; const float* skipB = nullptr;
    if (sW) {
        int stiles = cdiv(Cout, 16);
        k_pack_b<<<cdiv(kpad * stiles * 16, 256), 256, 0, s>>>(sW, nullptr, c.Bp2, Cin, kpad, Cout, 0, stiles);
        int stil = (Npad / 16) * stiles;
        k_wmma_gemm<<<cdiv(stil, 4), 128, 0, s>>>(c.Abf, c.Bp2, c.Sk, N, kpad, stiles);
        skipSrc = c.Sk; skipStride = stiles * 16; skipB = sb;
    } else {
        skipSrc = X; skipStride = Cin;  // identity skip (Cin == Cout)
    }
    // 7. finalize
    k_finalize<<<cdiv(N, 4), 128, 0, s>>>(c.acc, c.cnt, bias, g, be,
                                          skipSrc, skipStride, skipB, out, N, Cout, relu);
}

} // namespace

extern "C" void kernel_launch(void* const* d_in, const int* in_sizes, int n_in,
                              void* d_out, int out_size, void* d_ws, size_t ws_size,
                              hipStream_t stream) {
    (void)in_sizes; (void)n_in; (void)out_size; (void)ws_size;
    // --- inputs (setup_inputs dict order) ---
    const float* x    = (const float*)d_in[0];
    const int*   e0   = (const int*)d_in[1];
    const int*   e1   = (const int*)d_in[2];
    const int*   e2   = (const int*)d_in[3];
    const int*   cl1  = (const int*)d_in[4];
    const int*   cl2  = (const int*)d_in[5];
    const float* W01  = (const float*)d_in[9];
    const float* u01  = (const float*)d_in[10];
    const float* c01  = (const float*)d_in[11];
    const float* b01  = (const float*)d_in[12];
    const float* sW01 = (const float*)d_in[13];
    const float* sb01 = (const float*)d_in[14];
    const float* g01  = (const float*)d_in[15];
    const float* be01 = (const float*)d_in[16];
    const float* Ws   = (const float*)d_in[17];
    const float* us   = (const float*)d_in[18];
    const float* cs   = (const float*)d_in[19];
    const float* bs   = (const float*)d_in[20];
    const float* gs   = (const float*)d_in[21];
    const float* bes  = (const float*)d_in[22];
    const float* Wc   = (const float*)d_in[23];
    const float* uc   = (const float*)d_in[24];
    const float* ccv  = (const float*)d_in[25];
    const float* bcv  = (const float*)d_in[26];
    const float* sWc  = (const float*)d_in[27];
    const float* sbc  = (const float*)d_in[28];
    const float* gc   = (const float*)d_in[29];
    const float* bec  = (const float*)d_in[30];
    const float* Wo   = (const float*)d_in[31];
    const float* uo   = (const float*)d_in[32];
    const float* co   = (const float*)d_in[33];
    const float* bo   = (const float*)d_in[34];
    const float* sWo  = (const float*)d_in[35];
    const float* sbo  = (const float*)d_in[36];

    const int* e0s = e0, * e0d = e0 + E0;
    const int* e1s = e1, * e1d = e1 + E1;
    const int* e2s = e2, * e2d = e2 + E2;

    // --- workspace carving ---
    char* p = (char*)d_ws;
    auto carve = [&](size_t bytes) -> char* {
        char* r = p; p += (bytes + 255) & ~(size_t)255; return r;
    };
    const int N0p = 20000;  // multiple of 16
    Ctx c;
    c.stream = stream;
    c.Abf = (unsigned short*)carve((size_t)N0p * 288 * 2);       // bf16 activations (max 280->288)
    c.Bp  = (unsigned short*)carve((size_t)288 * 576 * 2);       // packed [W|u]
    c.Bp2 = (unsigned short*)carve((size_t)288 * 144 * 2);       // packed skip W
    c.M   = (float*)carve((size_t)N0p * 576 * 4);                // GEMM output
    c.Sk  = (float*)carve((size_t)N0p * 144 * 4);                // skip GEMM output
    c.acc = (float*)carve((size_t)N0p * Cc * 4);                 // edge/pool accum
    c.cnt = (float*)carve((size_t)N0p * 4);
    float* h0A   = (float*)carve((size_t)N0 * Cc * 4);
    float* h0B   = (float*)carve((size_t)N0 * Cc * 4);           // copy0
    float* h1A   = (float*)carve((size_t)N1 * Cc * 4);           // copy1
    float* h1B   = (float*)carve((size_t)N1 * Cc * 4);
    float* h2A   = (float*)carve((size_t)N2 * Cc * 4);
    float* hcat0 = (float*)carve((size_t)N0 * 2 * Cc * 4);
    float* hcat1 = (float*)carve((size_t)N1 * 2 * Cc * 4);

    auto same = [&](int i, float* X_, const int* es, const int* ed, int E, int N, float* out) {
        run_conv(c, X_, N, Cc, 160,
                 Ws + (size_t)i * Cc * Hh * Cc, us + (size_t)i * Cc * Hh,
                 cs + (size_t)i * Hh, bs + (size_t)i * Cc,
                 gs + (size_t)i * Cc, bes + (size_t)i * Cc,
                 Cc, nullptr, nullptr, es, ed, E, out, 1);
    };
    auto catblk = [&](int i, const float* X_, const int* es, const int* ed, int E, int N, float* out) {
        run_conv(c, X_, N, 2 * Cc, 288,
                 Wc + (size_t)i * 2 * Cc * Hh * Cc, uc + (size_t)i * 2 * Cc * Hh,
                 ccv + (size_t)i * Hh, bcv + (size_t)i * Cc,
                 gc + (size_t)i * Cc, bec + (size_t)i * Cc,
                 Cc, sWc + (size_t)i * 2 * Cc * Cc, sbc + (size_t)i * Cc,
                 es, ed, E, out, 1);
    };

    // conv01: 4 -> 140, LN, linear skip
    run_conv(c, x, N0, 4, 32, W01, u01, c01, b01, g01, be01,
             Cc, sW01, sb01, e0s, e0d, E0, h0A, 1);
    // conv02 (copy0 = h0B)
    same(0, h0A, e0s, e0d, E0, N0, h0B);
    // pool1: N0 -> N1
    hipMemsetAsync(c.acc, 0, (size_t)N1 * Cc * 4, stream);
    hipMemsetAsync(c.cnt, 0, (size_t)N1 * 4, stream);
    k_pool_acc<<<cdiv(N0 * Cc, 256), 256, 0, stream>>>(h0B, cl1, c.acc, c.cnt, N0, Cc);
    k_pool_div<<<cdiv(N1 * Cc, 256), 256, 0, stream>>>(c.acc, c.cnt, h1A, N1, Cc);
    // conv11, conv12 (in place; copy1 = h1A)
    same(1, h1A, e1s, e1d, E1, N1, h1A);
    same(2, h1A, e1s, e1d, E1, N1, h1A);
    // pool2: N1 -> N2
    hipMemsetAsync(c.acc, 0, (size_t)N2 * Cc * 4, stream);
    hipMemsetAsync(c.cnt, 0, (size_t)N2 * 4, stream);
    k_pool_acc<<<cdiv(N1 * Cc, 256), 256, 0, stream>>>(h1A, cl2, c.acc, c.cnt, N1, Cc);
    k_pool_div<<<cdiv(N2 * Cc, 256), 256, 0, stream>>>(c.acc, c.cnt, h2A, N2, Cc);
    // conv21, conv22 (in place)
    same(3, h2A, e2s, e2d, E2, N2, h2A);
    same(4, h2A, e2s, e2d, E2, N2, h2A);
    // unpool2 + concat -> conv13
    k_unpool_concat<<<cdiv(N1 * 2 * Cc, 256), 256, 0, stream>>>(h2A, cl2, h1A, hcat1, N1, Cc);
    catblk(0, hcat1, e1s, e1d, E1, N1, h1B);
    // conv14..16 (in place)
    same(5, h1B, e1s, e1d, E1, N1, h1B);
    same(6, h1B, e1s, e1d, E1, N1, h1B);
    same(7, h1B, e1s, e1d, E1, N1, h1B);
    // unpool1 + concat -> conv03
    k_unpool_concat<<<cdiv(N0 * 2 * Cc, 256), 256, 0, stream>>>(h1B, cl1, h0B, hcat0, N0, Cc);
    catblk(1, hcat0, e0s, e0d, E0, N0, h0A);
    // conv04, conv05 (in place)
    same(8, h0A, e0s, e0d, E0, N0, h0A);
    same(9, h0A, e0s, e0d, E0, N0, h0A);
    // conv06: 140 -> 1, no LN, no ReLU, linear skip -> d_out
    run_conv(c, h0A, N0, Cc, 160, Wo, uo, co, bo, nullptr, nullptr,
             1, sWo, sbo, e0s, e0d, E0, (float*)d_out, 0);
}
